// GaussMarkovBottleneck_56410100465706
// MI455X (gfx1250) — matrix operand order repaired and checked by reading, not compiled, not run on hardware
//
#include <hip/hip_runtime.h>
#include <cstdint>

#define B_  32
#define T_  2048
#define IN_ 512
#define L_  64
#define TR_ 128

typedef __attribute__((ext_vector_type(16))) __bf16 v16bf;
typedef __attribute__((ext_vector_type(8)))  float  v8f;

// ---------- workspace layout (bytes) ----------
#define OFF_WZH    0
#define OFF_WH2    (OFF_WZH   + 64*512*2)          // W_zh packed frags
#define OFF_WGP    (OFF_WH2   + 512*128*2)         // [W_hloc|W_hscale] packed
#define OFF_WGHPH  (OFF_WGP   + 64*256*2)          // [W_gz|W_pz] packed
#define OFF_WSIG   (OFF_WGHPH + 128*128*2)         // [W_gh|W_ph] packed
#define OFF_WLOC   (OFF_WSIG  + 64*64*2)           // W_sig packed
#define OFF_SCALES (OFF_WLOC  + 64*64*2)           // W_loc packed
#define OFF_KLSUM  (OFF_SCALES + (size_t)B_*T_*L_*4)

#define SCAN_LDS  233984
#define PRIOR_LDS 198656

__device__ __forceinline__ __bf16 f2bf(float f) {
  unsigned u = __float_as_uint(f);
  unsigned r = (u + 0x7FFFu + ((u >> 16) & 1u)) >> 16;   // round-to-nearest-even
  unsigned short h = (unsigned short)r;
  return __builtin_bit_cast(__bf16, h);
}

__device__ __forceinline__ v8f wmma_bf16(v16bf a, v16bf b, v8f c) {
  return __builtin_amdgcn_wmma_f32_16x16x32_bf16(false, a, false, b, (short)0, c,
                                                 false, false);
}

// ---- branchless transcendentals: v_exp/v_log/v_rcp only, no IEEE div sequences ----
__device__ __forceinline__ float fast_tanh(float x) {
  float xc = fminf(fmaxf(x, -10.f), 10.f);          // one v_med3_num_f32
  float e  = __expf(2.f * xc);
  return (e - 1.f) * __builtin_amdgcn_rcpf(e + 1.f);
}
__device__ __forceinline__ float fast_sigmoid(float x) {
  return __builtin_amdgcn_rcpf(1.f + __expf(-x));
}
__device__ __forceinline__ float fast_softplus(float x) {
  return fmaxf(x, 0.f) + __logf(1.f + __expf(-fabsf(x)));
}

// ---- fragment-ready layouts: each lane's 16 bf16 values are contiguous (32B) ----
// A-matrix 16x32 bf16 fragment (ISA 7.12.2):
//   lanes 0-15 : row=lane,    K = k0+{0..7,16..23}
//   lanes 16-31: row=lane-16, K = k0+{8..15,24..31}
__device__ __forceinline__ int afrag_idx(int r, int k) {
  int kt  = k >> 5, sub = k & 31;
  int lane = (r & 15) + ((sub & 8) ? 16 : 0);
  int idx  = ((sub & 16) ? 8 : 0) | (sub & 7);
  return (kt * 32 + lane) * 16 + idx;
}

__device__ __forceinline__ v16bf load_A_frag(const __bf16* Af, int kt, int lane) {
  const void* p = __builtin_assume_aligned(Af + (kt * 32 + lane) * 16, 32);
  return *(const v16bf*)p;
}

// B-matrix 32x16 fragment, packed per (kt, nt) tile as [lane][16]
__device__ __forceinline__ v16bf load_B_frag(const __bf16* Bp, int tiles_n,
                                             int kt, int nt, int lane) {
  const void* p = __builtin_assume_aligned(
      Bp + ((kt * tiles_n + nt) * 32 + lane) * 16, 32);
  return *(const v16bf*)p;
}

// decode packed-B destination index d -> source (k, n) for a K x N row-major matrix
__device__ __forceinline__ void bdec(int d, int N, int& k, int& n) {
  int idx = d & 15, lane = (d >> 4) & 31, tile = d >> 9;
  int tiles_n = N >> 4;
  int kt = tile / tiles_n, nt = tile - kt * tiles_n;
  k = kt * 32 + ((lane & 16) ? 8 : 0) + ((idx & 8) ? 16 : 0) + (idx & 7);
  n = nt * 16 + (lane & 15);
}

// ---------------- weight pack: f32 row-major -> bf16 fragment-ready ----------------
__global__ void pack_kernel(const float* Wzh, const float* Whl, const float* Whs,
                            const float* Wgz, const float* Wpz, const float* Wgh,
                            const float* Wph, const float* Wsig, const float* Wloc,
                            __bf16* dWzh, __bf16* dWh2, __bf16* dWgp,
                            __bf16* dWghph, __bf16* dWsig, __bf16* dWloc) {
  int d = blockIdx.x * blockDim.x + threadIdx.x;
  int k, n;
  if (d < 64 * 512) {                       // W_zh: K=64, N=512
    bdec(d, 512, k, n);
    dWzh[d] = f2bf(Wzh[k * 512 + n]);
  }
  if (d < 512 * 128) {                      // [W_hloc|W_hscale]: K=512, N=128
    bdec(d, 128, k, n);
    dWh2[d] = f2bf((n < 64) ? Whl[k * 64 + n] : Whs[k * 64 + (n - 64)]);
  }
  if (d < 64 * 256) {                       // [W_gz|W_pz]: K=64, N=256
    bdec(d, 256, k, n);
    dWgp[d] = f2bf((n < 128) ? Wgz[k * 128 + n] : Wpz[k * 128 + (n - 128)]);
  }
  if (d < 128 * 128) {                      // [W_gh|W_ph]: K=128, N=128
    bdec(d, 128, k, n);
    dWghph[d] = f2bf((n < 64) ? Wgh[k * 64 + n] : Wph[k * 64 + (n - 64)]);
  }
  if (d < 64 * 64) {                        // W_sig, W_loc: K=64, N=64
    bdec(d, 64, k, n);
    dWsig[d] = f2bf(Wsig[k * 64 + n]);
    dWloc[d] = f2bf(Wloc[k * 64 + n]);
  }
}

// ---------------- sequential posterior scan: 2 blocks x 16 batch rows ----------------
__global__ void __launch_bounds__(512)
scan_kernel(const float* __restrict__ x, const float* __restrict__ eps,
            const float* __restrict__ bzh, const float* __restrict__ bhl,
            const float* __restrict__ bhs, const float* __restrict__ zq0,
            const __bf16* __restrict__ Wzh, const __bf16* __restrict__ Wh2,
            float* __restrict__ zs_out, float* __restrict__ mus_out,
            float* __restrict__ scales_out) {
  extern __shared__ char smem[];
  __bf16* sWzh = (__bf16*)smem;                 // packed frags: 64*512
  __bf16* sWh2 = sWzh + 64 * 512;               // packed frags: 512*128
  __bf16* sZf  = sWh2 + 512 * 128;              // A-frags (2 kt): 1024
  __bf16* sHf  = sZf + 1024;                    // A-frags (16 kt): 8192
  float*  sGa  = (float*)(sHf + 8192);          // [16][128] split-K partial 0
  float*  sGb  = sGa + 16 * 128;                // [16][128] split-K partial 1
  float*  sbzh = sGb + 16 * 128;                // [512]
  float*  sbhl = sbzh + 512;                    // [64]
  float*  sbhs = sbhl + 64;                     // [64]

  const int tid  = threadIdx.x;
  const int lane = tid & 31;
  const int w    = tid >> 5;          // 16 waves
  const int blk  = blockIdx.x;        // batch rows blk*16 .. +15
  const int mb   = (lane >> 4) * 8;
  const int nn   = lane & 15;

  for (int i = tid; i < 64 * 512; i += 512)  sWzh[i] = Wzh[i];
  for (int i = tid; i < 512 * 128; i += 512) sWh2[i] = Wh2[i];
  if (tid < 512) sbzh[tid] = bzh[tid];
  if (tid < 64) { sbhl[tid] = bhl[tid]; sbhs[tid] = bhs[tid]; }
  for (int i = tid; i < 16 * 64; i += 512) {
    int r = i >> 6, c = i & 63;
    sZf[afrag_idx(r, c)] = f2bf(zq0[c]);
  }
  __syncthreads();

  for (int t = 0; t < T_; ++t) {
    // prefetch next step's x tile (one cacheline per thread)
    if (t + 1 < T_) {
      const float* px = &x[(((size_t)(blk * 16 + (tid >> 5)) * T_) + t + 1) * IN_ +
                           (tid & 31) * 16];
      __builtin_prefetch(px, 0, 1);
    }

    // GEMM1: H_pre = Z[16,64] @ W_zh[64,512]; h = 0.5*(tanh(H_pre+b)+x_t)
    v16bf a0 = load_A_frag(sZf, 0, lane);
    v16bf a1 = load_A_frag(sZf, 1, lane);
#pragma unroll
    for (int j = 0; j < 2; ++j) {
      int nt = w + j * 16;           // 32 N-tiles over 16 waves
      v8f acc = {};
      acc = wmma_bf16(a0, load_B_frag(sWzh, 32, 0, nt, lane), acc);
      acc = wmma_bf16(a1, load_B_frag(sWzh, 32, 1, nt, lane), acc);
      int col = nt * 16 + nn;
#pragma unroll
      for (int v = 0; v < 8; ++v) {
        int row = mb + v;
        float xv = x[(((size_t)(blk * 16 + row) * T_) + t) * IN_ + col];
        float hv = 0.5f * (fast_tanh(acc[v] + sbzh[col]) + xv);
        sHf[afrag_idx(row, col)] = f2bf(hv);
      }
    }
    __syncthreads();

    // GEMM2 split-K x2 over all 16 waves:
    //   wave w: n-tile (w&7), K-steps [(w>>3)*8, (w>>3)*8+8)
    {
      int nt2   = w & 7;
      int khalf = (w >> 3) * 8;
      v8f acc = {};
#pragma unroll
      for (int ks = 0; ks < 8; ++ks) {
        v16bf a = load_A_frag(sHf, khalf + ks, lane);
        v16bf b = load_B_frag(sWh2, 8, khalf + ks, nt2, lane);
        acc = wmma_bf16(a, b, acc);
      }
      int col = nt2 * 16 + nn;
      float* dst = (w >> 3) ? sGb : sGa;
#pragma unroll
      for (int v = 0; v < 8; ++v) dst[(mb + v) * 128 + col] = acc[v];
    }
    __syncthreads();

    // elementwise: combine split-K partials, z = mu + softplus(pre)*eps
    for (int i = tid; i < 16 * 64; i += 512) {
      int r = i >> 6, c = i & 63;
      int bg = blk * 16 + r;
      float mu = sGa[r * 128 + c] + sGb[r * 128 + c] + sbhl[c];
      float sp = sGa[r * 128 + 64 + c] + sGb[r * 128 + 64 + c] + sbhs[c];
      float sc = fast_softplus(sp);
      size_t idx = (((size_t)bg * T_) + t) * L_ + c;
      float z = mu + sc * eps[idx];
      zs_out[idx]     = z;
      mus_out[idx]    = mu;
      scales_out[idx] = sc;
      sZf[afrag_idx(r, c)] = f2bf(z);
    }
    __syncthreads();
  }
}

// ---------------- transition prior + per-(b,t) KL row sums ----------------
__global__ void __launch_bounds__(128)
prior_kernel(const float* __restrict__ zs, const float* __restrict__ mus,
             const float* __restrict__ scales,
             const __bf16* __restrict__ Wgp, const __bf16* __restrict__ Wghph,
             const __bf16* __restrict__ Wsig, const __bf16* __restrict__ Wloc,
             const float* __restrict__ bgz, const float* __restrict__ bpz,
             const float* __restrict__ bgh, const float* __restrict__ bph,
             const float* __restrict__ bsig, const float* __restrict__ bloc,
             float* __restrict__ klsum) {
  extern __shared__ char smem[];
  __bf16* sWgp   = (__bf16*)smem;         // packed: 64*256
  __bf16* sWghph = sWgp + 64 * 256;       // packed: 128*128
  __bf16* sWsig  = sWghph + 128 * 128;    // packed: 64*64
  __bf16* sWloc  = sWsig + 64 * 64;       // packed: 64*64
  float*  sbgp   = (float*)(sWloc + 64 * 64);  // [256]
  float*  sbghph = sbgp + 256;                 // [128]
  float*  sbsig  = sbghph + 128;               // [64]
  float*  sbloc  = sbsig + 64;                 // [64]
  char*   pw     = (char*)(sbloc + 64);

  const int tid  = threadIdx.x;
  const int lane = tid & 31;
  const int w    = tid >> 5;           // 4 waves
  const int mb   = (lane >> 4) * 8;
  const int nn   = lane & 15;

  const size_t per = 1024 * 2 + 4096 * 2 + 1024 * 2 + 4 * 1024 * 4;
  char* base = pw + (size_t)w * per;
  __bf16* sZtf  = (__bf16*)base;        // A-frags (2 kt)
  __bf16* sRf   = sZtf + 1024;          // A-frags (8 kt): relu(z@[Wgz|Wpz]+b)
  __bf16* sRpmf = sRf + 4096;           // A-frags (2 kt): relu(pm)
  float*  sGate = (float*)(sRpmf + 1024);  // [16][64]
  float*  sPm   = sGate + 16 * 64;         // [16][64]
  float*  sLoc  = sPm + 16 * 64;           // [16][64]
  float*  sSc   = sLoc + 16 * 64;          // [16][64]

  for (int i = tid; i < 64 * 256; i += 128)  sWgp[i] = Wgp[i];
  for (int i = tid; i < 128 * 128; i += 128) sWghph[i] = Wghph[i];
  for (int i = tid; i < 64 * 64; i += 128) { sWsig[i] = Wsig[i]; sWloc[i] = Wloc[i]; }
  for (int i = tid; i < 256; i += 128) sbgp[i] = (i < 128) ? bgz[i] : bpz[i - 128];
  if (tid < 128) sbghph[tid] = (tid < 64) ? bgh[tid] : bph[tid - 64];
  if (tid < 64) { sbsig[tid] = bsig[tid]; sbloc[tid] = bloc[tid]; }
  __syncthreads();

  int tileId = blockIdx.x * 4 + w;     // 0..4095
  int b      = tileId >> 7;            // 128 t-tiles per batch row
  int tbase  = (tileId & 127) << 4;

  // load 16 rows of z (f32 -> bf16 A-frags)
  for (int i = lane; i < 16 * 64; i += 32) {
    int r = i >> 6, c = i & 63;
    sZtf[afrag_idx(r, c)] = f2bf(zs[(((size_t)b * T_) + tbase + r) * L_ + c]);
  }

  v16bf a0 = load_A_frag(sZtf, 0, lane);
  v16bf a1 = load_A_frag(sZtf, 1, lane);

  // step1: r = relu(z @ [Wgz|Wpz] + [bgz|bpz])   [16,256]
  for (int nt = 0; nt < 16; ++nt) {
    v8f acc = {};
    acc = wmma_bf16(a0, load_B_frag(sWgp, 16, 0, nt, lane), acc);
    acc = wmma_bf16(a1, load_B_frag(sWgp, 16, 1, nt, lane), acc);
    int col = nt * 16 + nn;
#pragma unroll
    for (int v = 0; v < 8; ++v)
      sRf[afrag_idx(mb + v, col)] = f2bf(fmaxf(acc[v] + sbgp[col], 0.f));
  }

  // step2: gate = sigmoid(r_g @ Wgh + bgh), pm = r_p @ Wph + bph
  for (int half = 0; half < 2; ++half) {
    for (int nt = 0; nt < 4; ++nt) {
      v8f acc = {};
#pragma unroll
      for (int ks = 0; ks < 4; ++ks) {
        v16bf a  = load_A_frag(sRf, half * 4 + ks, lane);
        v16bf bb = load_B_frag(sWghph, 8, ks, half * 4 + nt, lane);
        acc = wmma_bf16(a, bb, acc);
      }
      int col = nt * 16 + nn;
      float bias = sbghph[half * 64 + col];
#pragma unroll
      for (int v = 0; v < 8; ++v) {
        int r = mb + v;
        float val = acc[v] + bias;
        if (half == 0) {
          sGate[r * 64 + col] = fast_sigmoid(val);
        } else {
          sPm[r * 64 + col]   = val;
          sRpmf[afrag_idx(r, col)] = f2bf(fmaxf(val, 0.f));
        }
      }
    }
  }

  // step3a: loc = (1-gate)*(z@Wloc + bloc) + gate*pm
  for (int nt = 0; nt < 4; ++nt) {
    v8f acc = {};
    acc = wmma_bf16(a0, load_B_frag(sWloc, 4, 0, nt, lane), acc);
    acc = wmma_bf16(a1, load_B_frag(sWloc, 4, 1, nt, lane), acc);
    int col = nt * 16 + nn;
#pragma unroll
    for (int v = 0; v < 8; ++v) {
      int r = mb + v;
      float zl = acc[v] + sbloc[col];
      float g  = sGate[r * 64 + col];
      sLoc[r * 64 + col] = (1.f - g) * zl + g * sPm[r * 64 + col];
    }
  }

  // step3b: sc = softplus(relu(pm) @ Wsig + bsig)
  {
    v16bf p0 = load_A_frag(sRpmf, 0, lane);
    v16bf p1 = load_A_frag(sRpmf, 1, lane);
    for (int nt = 0; nt < 4; ++nt) {
      v8f acc = {};
      acc = wmma_bf16(p0, load_B_frag(sWsig, 4, 0, nt, lane), acc);
      acc = wmma_bf16(p1, load_B_frag(sWsig, 4, 1, nt, lane), acc);
      int col = nt * 16 + nn;
#pragma unroll
      for (int v = 0; v < 8; ++v)
        sSc[(mb + v) * 64 + col] = fast_softplus(acc[v] + sbsig[col]);
    }
  }

  // step4: KL row sums. Prior from time gt applies at time gt+1 (one writer each).
  for (int r = 0; r < 16; ++r) {
    int gt = tbase + r;
    int tp = gt + 1;
    float lsum = 0.f;
    if (tp < T_) {
#pragma unroll
      for (int j = 0; j < 2; ++j) {
        int c = lane * 2 + j;
        size_t idx = (((size_t)b * T_) + tp) * L_ + c;
        float mq = mus[idx], sq = scales[idx];
        float vq = sq * sq;
        float sp = sSc[r * 64 + c];
        float vp  = sp * sp;
        float rvp = __builtin_amdgcn_rcpf(vp);
        float d   = sLoc[r * 64 + c] - mq;
        lsum += 0.5f * (__logf(vp) - __logf(vq) - 1.f + vq * rvp + d * d * rvp);
      }
    }
    for (int o = 16; o > 0; o >>= 1) lsum += __shfl_xor(lsum, o, 32);
    if (lane == 0 && tp < T_) klsum[(size_t)b * T_ + tp] = lsum;
  }
  if (tbase == 0) {  // t=0 prior is N(0,1)
    float lsum = 0.f;
#pragma unroll
    for (int j = 0; j < 2; ++j) {
      int c = lane * 2 + j;
      size_t idx = ((size_t)b * T_) * L_ + c;
      float mq = mus[idx], sq = scales[idx];
      float vq = sq * sq;
      lsum += 0.5f * (-__logf(vq) - 1.f + vq + mq * mq);
    }
    for (int o = 16; o > 0; o >>= 1) lsum += __shfl_xor(lsum, o, 32);
    if (lane == 0) klsum[(size_t)b * T_] = lsum;
  }
}

// ---------------- deterministic per-batch reduction over T ----------------
__global__ void reduce_kernel(const float* __restrict__ klsum, float* __restrict__ loss) {
  __shared__ float red[256];
  int b = blockIdx.x;
  float s = 0.f;
  for (int t = threadIdx.x; t < T_; t += 256) s += klsum[(size_t)b * T_ + t];
  red[threadIdx.x] = s;
  __syncthreads();
  for (int o = 128; o > 0; o >>= 1) {
    if (threadIdx.x < o) red[threadIdx.x] += red[threadIdx.x + o];
    __syncthreads();
  }
  if (threadIdx.x == 0) loss[b] = red[0];
}

extern "C" void kernel_launch(void* const* d_in, const int* in_sizes, int n_in,
                              void* d_out, int out_size, void* d_ws, size_t ws_size,
                              hipStream_t stream) {
  const float* x    = (const float*)d_in[0];
  const float* eps  = (const float*)d_in[1];
  const float* Wzh  = (const float*)d_in[2];
  const float* bzh  = (const float*)d_in[3];
  const float* Whl  = (const float*)d_in[4];
  const float* bhl  = (const float*)d_in[5];
  const float* Whs  = (const float*)d_in[6];
  const float* bhs  = (const float*)d_in[7];
  const float* Wgz  = (const float*)d_in[8];
  const float* bgz  = (const float*)d_in[9];
  const float* Wgh  = (const float*)d_in[10];
  const float* bgh  = (const float*)d_in[11];
  const float* Wpz  = (const float*)d_in[12];
  const float* bpz  = (const float*)d_in[13];
  const float* Wph  = (const float*)d_in[14];
  const float* bph  = (const float*)d_in[15];
  const float* Wsig = (const float*)d_in[16];
  const float* bsig = (const float*)d_in[17];
  const float* Wloc = (const float*)d_in[18];
  const float* bloc = (const float*)d_in[19];
  const float* zq0  = (const float*)d_in[20];

  char* ws = (char*)d_ws;
  __bf16* dWzh   = (__bf16*)(ws + OFF_WZH);
  __bf16* dWh2   = (__bf16*)(ws + OFF_WH2);
  __bf16* dWgp   = (__bf16*)(ws + OFF_WGP);
  __bf16* dWghph = (__bf16*)(ws + OFF_WGHPH);
  __bf16* dWsig  = (__bf16*)(ws + OFF_WSIG);
  __bf16* dWloc  = (__bf16*)(ws + OFF_WLOC);
  float*  scales = (float*)(ws + OFF_SCALES);
  float*  klsum  = (float*)(ws + OFF_KLSUM);

  float* out = (float*)d_out;
  const size_t BTL = (size_t)B_ * T_ * L_;
  float* zs_out   = out;
  float* loss_out = out + BTL;
  float* mus_out  = out + BTL + B_;

  pack_kernel<<<256, 256, 0, stream>>>(Wzh, Whl, Whs, Wgz, Wpz, Wgh, Wph, Wsig, Wloc,
                                       dWzh, dWh2, dWgp, dWghph, dWsig, dWloc);

  hipFuncSetAttribute(reinterpret_cast<const void*>(scan_kernel),
                      hipFuncAttributeMaxDynamicSharedMemorySize, SCAN_LDS);
  scan_kernel<<<2, 512, SCAN_LDS, stream>>>(x, eps, bzh, bhl, bhs, zq0, dWzh, dWh2,
                                            zs_out, mus_out, scales);

  hipFuncSetAttribute(reinterpret_cast<const void*>(prior_kernel),
                      hipFuncAttributeMaxDynamicSharedMemorySize, PRIOR_LDS);
  prior_kernel<<<1024, 128, PRIOR_LDS, stream>>>(zs_out, mus_out, scales,
                                                 dWgp, dWghph, dWsig, dWloc,
                                                 bgz, bpz, bgh, bph, bsig, bloc, klsum);

  reduce_kernel<<<B_, 256, 0, stream>>>(klsum, loss_out);
}
